// RNNSeq2Seq_38070590112382
// MI455X (gfx1250) — compile-verified, matrix operand order checked by reading, not run on hardware
//
#include <hip/hip_runtime.h>
#include <hip/hip_bf16.h>
#include <stdint.h>

// ---------------------------------------------------------------------------
// Model dims (fixed by the reference)
// ---------------------------------------------------------------------------
#define LYR 2
#define HID 1024
#define EMB 512
#define VT  32000
#define SLEN 64
#define TLEN 64
#define BAT 64
#define G4H (4 * HID)   // 4096
#define LPAD 8          // LDS row pad (halves): keeps row stride == 4 dwords mod 64 banks

typedef __bf16 v16bf __attribute__((ext_vector_type(16)));
typedef __bf16 v8bf  __attribute__((ext_vector_type(8)));
typedef float  v8f   __attribute__((ext_vector_type(8)));

// ---------------------------------------------------------------------------
// f32 -> bf16 conversion (grid-stride)
// ---------------------------------------------------------------------------
__global__ void cvt_bf16_kernel(const float* __restrict__ src,
                                __bf16* __restrict__ dst, size_t n) {
    size_t i = (size_t)blockIdx.x * blockDim.x + threadIdx.x;
    size_t stride = (size_t)gridDim.x * blockDim.x;
    for (; i < n; i += stride) dst[i] = (__bf16)src[i];
}

// ---------------------------------------------------------------------------
// CDNA5 async global->LDS copy (16B per lane), tracked by ASYNCcnt.
//   lds_off: wave-relative LDS byte offset (low 32 bits of shared pointer)
// ---------------------------------------------------------------------------
__device__ __forceinline__ void async_copy16(uint32_t lds_off, const void* gptr) {
    asm volatile("global_load_async_to_lds_b128 %0, %1, off"
                 :
                 : "v"(lds_off), "v"((uint64_t)(uintptr_t)gptr)
                 : "memory");
}
__device__ __forceinline__ void wait_async0() {
    asm volatile("s_wait_asynccnt 0x0" ::: "memory");
}

// ---------------------------------------------------------------------------
// Fragment loaders: 16 bf16 per lane as two 16B loads per the CDNA5 ISA layout
//   lanes 0-15 hold k = {0..7, 16..23}, lanes 16-31 hold k = {8..15, 24..31}
//   (caller pre-offsets the pointer by ((lane>>4)<<3))
// ---------------------------------------------------------------------------
__device__ __forceinline__ v16bf load_frag(const __bf16* p) {
    v8bf lo = *reinterpret_cast<const v8bf*>(p);
    v8bf hi = *reinterpret_cast<const v8bf*>(p + 16);
    return __builtin_shufflevector(lo, hi, 0, 1, 2, 3, 4, 5, 6, 7,
                                   8, 9, 10, 11, 12, 13, 14, 15);
}

// ---------------------------------------------------------------------------
// Fused dual-GEMM + bias, bf16 WMMA with f32 accumulation:
//   C[64, N] = X[64,Kx] * Wx[N,Kx]^T  (+ Hs[64,Kh] * Wh[N,Kh]^T) (+ bias0 + bias1)
//
// Block = 128 threads = 4 waves; wave w owns column tile ntile = blockIdx*4+w
// and all 4 M-tiles (M=64):
//   - A (activations) staged once per block into LDS via async DMA copies,
//     shared by all 4 waves, read back as conflict-free ds_load_b128 fragments
//     double-buffered one k-step ahead of the WMMAs.
//   - B (weights) streamed from global/L2, also pipelined one k-step ahead.
// grid.x = N / 64.  Dynamic LDS: 64*((Kx+LPAD) [+ (Kh+LPAD)]) bf16.
// ---------------------------------------------------------------------------
__global__ __launch_bounds__(128) void gemm_bias_kernel(
    const __bf16* __restrict__ X, int Kx, const __bf16* __restrict__ Wx,
    const __bf16* __restrict__ Hs, int Kh, const __bf16* __restrict__ Wh,
    const float* __restrict__ bias0, const float* __restrict__ bias1,
    float* __restrict__ C, int ldc) {
    extern __shared__ __bf16 lds[];

    const int tid   = threadIdx.x;
    const int lane  = tid & 31;
    const int wave  = tid >> 5;
    const int ntile = blockIdx.x * 4 + wave;
    const int lm    = lane & 15;
    const int koff  = (lane >> 4) << 3;   // 0 or 8 (halves)
    const int ncol  = ntile * 16 + lm;

    const int ldx = Kx + LPAD;
    const int ldh = Kh + LPAD;
    __bf16* lX = lds;
    __bf16* lH = lds + 64 * ldx;
    const uint32_t lX_off = (uint32_t)(uintptr_t)(void*)lX;   // low 32 bits = LDS byte address
    const uint32_t lH_off = (uint32_t)(uintptr_t)(void*)lH;

    // ---- async DMA stage of A matrices into LDS (16B per lane, ASYNCcnt) ----
    {
        int nv = 64 * (Kx >> 3);                 // 8-half (16B) vectors
        for (int i = tid; i < nv; i += 128) {
            int row = i / (Kx >> 3), c8 = i - row * (Kx >> 3);
            async_copy16(lX_off + (uint32_t)(row * ldx + c8 * 8) * 2u,
                         X + (size_t)row * Kx + c8 * 8);
        }
        if (Hs != nullptr) {
            nv = 64 * (Kh >> 3);
            for (int i = tid; i < nv; i += 128) {
                int row = i / (Kh >> 3), c8 = i - row * (Kh >> 3);
                async_copy16(lH_off + (uint32_t)(row * ldh + c8 * 8) * 2u,
                             Hs + (size_t)row * Kh + c8 * 8);
            }
        }
    }
    wait_async0();
    __syncthreads();

    v8f acc[4];
#pragma unroll
    for (int m = 0; m < 4; ++m) acc[m] = (v8f)(0.0f);

    // One GEMM segment: both the global weight stream (B) and the LDS
    // activation fragments (A) are pipelined one k-step ahead of the WMMAs.
    auto segment = [&](const __bf16* Wrow, const __bf16* lA, int K, int ldl) {
        const __bf16* wp = Wrow + koff;
        const __bf16* ap = lA + (size_t)lm * ldl + koff;
        v16bf b = load_frag(wp);
        v16bf a[4];
#pragma unroll
        for (int m = 0; m < 4; ++m)
            a[m] = load_frag(ap + (size_t)(m * 16) * ldl);
        int k = 0;
        for (; k + 32 < K; k += 32) {
            v16bf bn = load_frag(wp + k + 32);           // next weight frag
            v16bf an[4];
#pragma unroll
            for (int m = 0; m < 4; ++m)                  // next A frags (ds)
                an[m] = load_frag(ap + (size_t)(m * 16) * ldl + k + 32);
#pragma unroll
            for (int m = 0; m < 4; ++m)
                acc[m] = __builtin_amdgcn_wmma_f32_16x16x32_bf16(
                    false, a[m], false, b, (short)0, acc[m], false, false);
#pragma unroll
            for (int m = 0; m < 4; ++m) a[m] = an[m];
            b = bn;
        }
#pragma unroll
        for (int m = 0; m < 4; ++m)                      // epilogue k-step
            acc[m] = __builtin_amdgcn_wmma_f32_16x16x32_bf16(
                false, a[m], false, b, (short)0, acc[m], false, false);
    };

    segment(Wx + (size_t)ncol * Kx, lX, Kx, ldx);
    if (Hs != nullptr) segment(Wh + (size_t)ncol * Kh, lH, Kh, ldh);

    float bias = bias0 ? bias0[ncol] : 0.0f;
    if (bias1) bias += bias1[ncol];

    // C/D layout: lane holds column ncol; rows r + (lane>=16 ? 8 : 0), VGPR r
    const int mbase = (lane >> 4) << 3;
#pragma unroll
    for (int m = 0; m < 4; ++m) {
#pragma unroll
        for (int r = 0; r < 8; ++r) {
            C[(size_t)(m * 16 + mbase + r) * ldc + ncol] = acc[m][r] + bias;
        }
    }
}

// ---------------------------------------------------------------------------
// LSTM gate activation: g = [i | f | g | o] rows of 4096 per batch element.
// ---------------------------------------------------------------------------
__device__ __forceinline__ float fsig(float x) {
    return 1.0f / (1.0f + __expf(-x));
}
__device__ __forceinline__ float ftanh(float x) {
    return 2.0f * fsig(2.0f * x) - 1.0f;
}

__global__ __launch_bounds__(256) void lstm_act_kernel(
    const float* __restrict__ G, float* __restrict__ c,
    __bf16* __restrict__ h, __bf16* __restrict__ xout) {
    int idx = blockIdx.x * blockDim.x + threadIdx.x;   // B*H = 65536
    int b = idx >> 10;
    int j = idx & (HID - 1);
    const float* row = G + (size_t)b * G4H;
    float gi = fsig(row[j]);
    float gf = fsig(row[HID + j]);
    float gg = ftanh(row[2 * HID + j]);
    float go = fsig(row[3 * HID + j]);
    float cn = gf * c[idx] + gi * gg;
    c[idx] = cn;
    __bf16 hv = (__bf16)(go * ftanh(cn));
    h[idx] = hv;
    if (xout) xout[idx] = hv;
}

// ---------------------------------------------------------------------------
// Embedding gather (f32 table -> bf16 activations)
// ---------------------------------------------------------------------------
__global__ void gather_emb_kernel(const int* __restrict__ tok,
                                  const float* __restrict__ emb,
                                  __bf16* __restrict__ out,
                                  int E, int rows) {
    int idx = blockIdx.x * blockDim.x + threadIdx.x;
    if (idx >= rows * E) return;
    int r = idx / E;
    int e = idx - r * E;
    out[idx] = (__bf16)emb[(size_t)tok[r] * E + e];
}

__global__ void copy_tok_kernel(const int* __restrict__ src,
                                int* __restrict__ tok) {
    tok[threadIdx.x] = src[threadIdx.x];
}

// ---------------------------------------------------------------------------
// Row argmax over VT with first-max tie rule (matches jnp.argmax)
// ---------------------------------------------------------------------------
__global__ __launch_bounds__(256) void argmax_row_kernel(
    const float* __restrict__ logits, int* __restrict__ tok, int N) {
    int b = blockIdx.x;
    const float* row = logits + (size_t)b * N;
    float best = -__builtin_inff();
    int bi = 0;
    for (int j = threadIdx.x; j < N; j += 256) {
        float v = row[j];
        if (v > best) { best = v; bi = j; }
    }
    __shared__ float sv[256];
    __shared__ int   si[256];
    sv[threadIdx.x] = best;
    si[threadIdx.x] = bi;
    __syncthreads();
    for (int s = 128; s > 0; s >>= 1) {
        if (threadIdx.x < s) {
            float v2 = sv[threadIdx.x + s];
            int   i2 = si[threadIdx.x + s];
            if (v2 > sv[threadIdx.x] ||
                (v2 == sv[threadIdx.x] && i2 < si[threadIdx.x])) {
                sv[threadIdx.x] = v2;
                si[threadIdx.x] = i2;
            }
        }
        __syncthreads();
    }
    if (threadIdx.x == 0) tok[b] = si[0];
}

// ---------------------------------------------------------------------------
// Host-side orchestration
// ---------------------------------------------------------------------------
static inline size_t align256(size_t x) { return (x + 255) & ~(size_t)255; }
static inline size_t lds_bytes(int Kx, int Kh /*0 if none*/) {
    size_t halves = (size_t)64 * (Kx + LPAD);
    if (Kh) halves += (size_t)64 * (Kh + LPAD);
    return halves * 2;
}

extern "C" void kernel_launch(void* const* d_in, const int* in_sizes, int n_in,
                              void* d_out, int out_size, void* d_ws, size_t ws_size,
                              hipStream_t stream) {
    (void)in_sizes; (void)n_in; (void)out_size;

    // Input order per setup_inputs()
    const int*   src     = (const int*)d_in[0];    // [S, B]
    const int*   trg     = (const int*)d_in[1];    // [T, B]
    const float* src_emb = (const float*)d_in[2];  // [VS, E]
    const float* trg_emb = (const float*)d_in[3];  // [VT, E]
    const float* W_f32[8];   // enc/dec Wih0,Whh0,Wih1,Whh1
    const float* bih[4];     // enc0, enc1, dec0, dec1
    const float* bhh[4];
    {
        int k = 4;
        for (int p = 0; p < 2; ++p)
            for (int l = 0; l < 2; ++l) {
                W_f32[p * 4 + l * 2 + 0] = (const float*)d_in[k++]; // Wih
                W_f32[p * 4 + l * 2 + 1] = (const float*)d_in[k++]; // Whh
                bih[p * 2 + l] = (const float*)d_in[k++];
                bhh[p * 2 + l] = (const float*)d_in[k++];
            }
    }
    const float* Wout = (const float*)d_in[20];
    const float* bout = (const float*)d_in[21];
    float* out = (float*)d_out;

    // ---- carve workspace ----
    char* wp = (char*)d_ws;
    auto carve = [&](size_t bytes) -> void* {
        void* p = (void*)wp;
        wp += align256(bytes);
        return p;
    };
    const size_t wsz[8] = {
        (size_t)G4H * EMB, (size_t)G4H * HID, (size_t)G4H * HID, (size_t)G4H * HID,
        (size_t)G4H * EMB, (size_t)G4H * HID, (size_t)G4H * HID, (size_t)G4H * HID };
    __bf16* W_bf[8];
    for (int i = 0; i < 8; ++i) W_bf[i] = (__bf16*)carve(wsz[i] * 2);
    __bf16* Wout_bf = (__bf16*)carve((size_t)VT * HID * 2);
    __bf16* xseq0   = (__bf16*)carve((size_t)SLEN * BAT * EMB * 2);
    __bf16* xseq1   = (__bf16*)carve((size_t)SLEN * BAT * HID * 2);
    __bf16* hstate  = (__bf16*)carve((size_t)LYR * BAT * HID * 2);
    float*  cstate  = (float*)carve((size_t)LYR * BAT * HID * 4);
    float*  Gbuf    = (float*)carve((size_t)BAT * G4H * 4);
    __bf16* xdec    = (__bf16*)carve((size_t)BAT * EMB * 2);
    int*    tok     = (int*)carve(BAT * 4);
    if ((size_t)(wp - (char*)d_ws) > ws_size) return;  // workspace too small

    __bf16* h0 = hstate;
    __bf16* h1 = hstate + (size_t)BAT * HID;
    float*  c0 = cstate;
    float*  c1 = cstate + (size_t)BAT * HID;

    // ---- one-shot weight conversion to bf16 (L2-resident working set) ----
    for (int i = 0; i < 8; ++i)
        cvt_bf16_kernel<<<2048, 256, 0, stream>>>(W_f32[i], W_bf[i], wsz[i]);
    cvt_bf16_kernel<<<4096, 256, 0, stream>>>(Wout, Wout_bf, (size_t)VT * HID);

    // ---- encoder ----
    {
        int n = SLEN * BAT * EMB;
        gather_emb_kernel<<<(n + 255) / 256, 256, 0, stream>>>(
            src, src_emb, xseq0, EMB, SLEN * BAT);
    }
    hipMemsetAsync(hstate, 0, (size_t)LYR * BAT * HID * 2, stream);
    hipMemsetAsync(cstate, 0, (size_t)LYR * BAT * HID * 4, stream);

    // layer 0 scan (input dim E), emits xseq1 for layer 1
    for (int t = 0; t < SLEN; ++t) {
        gemm_bias_kernel<<<G4H / 64, 128, lds_bytes(EMB, HID), stream>>>(
            xseq0 + (size_t)t * BAT * EMB, EMB, W_bf[0],
            h0, HID, W_bf[1], bih[0], bhh[0], Gbuf, G4H);
        lstm_act_kernel<<<BAT * HID / 256, 256, 0, stream>>>(
            Gbuf, c0, h0, xseq1 + (size_t)t * BAT * HID);
    }
    // layer 1 scan (input dim H)
    for (int t = 0; t < SLEN; ++t) {
        gemm_bias_kernel<<<G4H / 64, 128, lds_bytes(HID, HID), stream>>>(
            xseq1 + (size_t)t * BAT * HID, HID, W_bf[2],
            h1, HID, W_bf[3], bih[1], bhh[1], Gbuf, G4H);
        lstm_act_kernel<<<BAT * HID / 256, 256, 0, stream>>>(
            Gbuf, c1, h1, nullptr);
    }

    // ---- decoder (greedy, 63 steps; step 0 of output stays zero) ----
    copy_tok_kernel<<<1, BAT, 0, stream>>>(trg, tok);
    hipMemsetAsync(out, 0, (size_t)BAT * VT * sizeof(float), stream);

    for (int t = 1; t < TLEN; ++t) {
        int n = BAT * EMB;
        gather_emb_kernel<<<(n + 255) / 256, 256, 0, stream>>>(
            tok, trg_emb, xdec, EMB, BAT);

        gemm_bias_kernel<<<G4H / 64, 128, lds_bytes(EMB, HID), stream>>>(
            xdec, EMB, W_bf[4], h0, HID, W_bf[5], bih[2], bhh[2], Gbuf, G4H);
        lstm_act_kernel<<<BAT * HID / 256, 256, 0, stream>>>(
            Gbuf, c0, h0, nullptr);

        gemm_bias_kernel<<<G4H / 64, 128, lds_bytes(HID, HID), stream>>>(
            h0, HID, W_bf[6], h1, HID, W_bf[7], bih[3], bhh[3], Gbuf, G4H);
        lstm_act_kernel<<<BAT * HID / 256, 256, 0, stream>>>(
            Gbuf, c1, h1, nullptr);

        float* logits = out + (size_t)t * BAT * VT;
        gemm_bias_kernel<<<VT / 64, 128, lds_bytes(HID, 0), stream>>>(
            h1, HID, Wout_bf, nullptr, 0, nullptr, bout, nullptr, logits, VT);
        argmax_row_kernel<<<BAT, 256, 0, stream>>>(logits, tok, VT);
    }
}
